// GenLSTMpdt_9938554323549
// MI455X (gfx1250) — matrix operand order repaired and checked by reading, not compiled, not run on hardware
//
#include <hip/hip_runtime.h>
#include <hip/hip_bf16.h>

// Problem constants (from reference)
#define B_   512
#define T_   256
#define SD_  8
#define ND_  32
#define H_   256
#define DT_  0.01f

// Tiling: 32 batch rows per block (2 WMMA M-tiles), 16 blocks, 8 waves/block.
#define MT_   2
#define ROWS_ 32

typedef __attribute__((ext_vector_type(16))) _Float16 v16h;
typedef __attribute__((ext_vector_type(8)))  _Float16 v8h;
typedef __attribute__((ext_vector_type(8)))  float    v8f;

// Global (address_space(1)) views of the packed fragment buffer: guarantees
// global_load_b128 (LOADcnt only) instead of flat_load_b128 (LOAD+DScnt).
typedef __attribute__((address_space(1))) const _Float16 gcHalf;
typedef __attribute__((address_space(1))) const v16h    gcv16h;

// ---- packed-weight fragment offsets (units of 512-half fragments) ----
// Fragment = one 16x16x32 WMMA B operand: 32 lanes x 16 halfs (1 KB).
// Order within a weight: frag = nt * nKc + kc.
#define OFF_WX   0      // Wx  [40 ->pad64, 1024]  : 2 kc x 64 nt = 128
#define OFF_WH   128    // Wh  [256, 1024]         : 8 kc x 64 nt = 512
#define OFF_WM1  640    // Wm1 [256, 256]          : 8 x 16 = 128
#define OFF_WM2  768
#define OFF_WM3  896    // Wm3 [256, 8->pad16]     : 8 x 1 = 8
#define OFF_WV1  904
#define OFF_WV2  1032
#define OFF_WV3  1160
#define N_FRAGS  1168   // ~1.17 MB f16 in workspace -> L2 resident

// ---------------------------------------------------------------------------
// Prep: f32 weights -> f16 WMMA B-matrix fragments.
//   B 32x16 f16 layout (wave32): N = lane%16; lanes 0-15 hold K {0..7,16..23}, lanes 16-31 hold
//   K {8..15,24..31}; VGPR v holds the f16 pair (K=k, K=k+1), k = 8*(lane/16) + 2v + 8*(v/4).
// ---------------------------------------------------------------------------
__global__ void pack_b_frags(const float* __restrict__ W, int K, int N,
                             int nKc, int nNt, unsigned* __restrict__ dst) {
  int tid = blockIdx.x * blockDim.x + threadIdx.x;
  int total = nKc * nNt * 256;
  if (tid >= total) return;
  int f    = tid >> 8;
  int r    = tid & 255;
  int lane = r >> 3;
  int v    = r & 7;
  int nt   = f / nKc;
  int kc   = f - nt * nKc;
  int k    = kc * 32 + ((lane >> 4) << 3) + 2 * v + ((v >> 2) << 3);
  int n0   = nt * 16 + (lane & 15);
  _Float16 h0 = (k     < K && n0 < N) ? (_Float16)W[(size_t)k * N + n0]       : (_Float16)0.f;
  _Float16 h1 = (k + 1 < K && n0 < N) ? (_Float16)W[(size_t)(k + 1) * N + n0] : (_Float16)0.f;
  union { _Float16 h[2]; unsigned u; } pk;
  pk.h[0] = h0; pk.h[1] = h1;
  dst[(size_t)f * 256 + lane * 8 + v] = pk.u;
}

// ---------------------------------------------------------------------------
// Device helpers
// ---------------------------------------------------------------------------
__device__ __forceinline__ v16h load_a(const _Float16* base, int stride, int kc,
                                       int lane) {
  // 16-bit A 16x32 layout: lanes 0-15 -> K {0..7, 16..23}, lanes 16-31 -> +8.
  int m  = lane & 15;
  int k0 = kc * 32 + (lane >> 4) * 8;
  const _Float16* p = base + m * stride + k0;
  union { v16h v; v8h h[2]; } u;
  u.h[0] = *(const v8h*)(p);
  u.h[1] = *(const v8h*)(p + 16);
  return u.v;
}

__device__ __forceinline__ v16h load_b(gcHalf* frags, int fragIdx, int lane) {
  return *(gcv16h*)(frags + (size_t)fragIdx * 512 + lane * 16);
}

__device__ __forceinline__ v8f wmma16(v16h a, v16h b, v8f c) {
  return __builtin_amdgcn_wmma_f32_16x16x32_f16(false, a, false, b, (short)0, c,
                                                false, false);
}

__device__ __forceinline__ float sigm_(float x) {
  return 1.f / (1.f + __expf(-x));
}
__device__ __forceinline__ float tanh_(float x) {
  return 1.f - 2.f / (__expf(2.f * x) + 1.f);
}

// ---------------------------------------------------------------------------
// Main persistent scan kernel: 16 blocks x 32 batch rows (2 M-tiles),
// 8 waves / block. Recurrent state stays in LDS / VGPRs; weight fragments
// stream from L2 each step. The fragment base is laundered as an integer in
// an SGPR per step (keeps loads inside the t-loop, no hoist/spill) and then
// cast to an address_space(1) pointer (forces global_load_b128 form).
// B fragments are software-pipelined one K-chunk ahead of the WMMAs.
// ---------------------------------------------------------------------------
__global__ __launch_bounds__(256) void genlstm_seq(
    const float* __restrict__ noise, const float* __restrict__ eps,
    const float* __restrict__ b4,  const float* __restrict__ bm1,
    const float* __restrict__ bm2, const float* __restrict__ bm3,
    const float* __restrict__ bv1, const float* __restrict__ bv2,
    const float* __restrict__ bv3, const _Float16* __restrict__ frags,
    float* __restrict__ out) {
  __shared__ _Float16 s_xt[ROWS_ * 64];          // [x | noise | pad] f16
  __shared__ _Float16 s_h[2][ROWS_ * 256];       // h ping-pong (t-1 / t)
  __shared__ _Float16 s_a1[2][ROWS_ * 256];      // layer-1 act (mu / var)
  __shared__ _Float16 s_a2[2][ROWS_ * 256];      // layer-2 act
  __shared__ float    s_mu[ROWS_ * SD_], s_lv[ROWS_ * SD_];
  __shared__ float    s_x[ROWS_ * SD_],  s_cum[ROWS_ * SD_];

  const int tid   = threadIdx.x;
  const int lane  = tid & 31;
  const int w     = tid >> 5;       // wave 0..7
  const int bm    = lane & 15;      // C/D column within tile; A row
  const int ch    = lane >> 4;      // half select (C/D rows +8)
  const int brow0 = blockIdx.x * ROWS_;

  // ---- init state ----
  for (int i = tid; i < ROWS_ * 256; i += 256) s_h[0][i] = (_Float16)0.f;
  {
    int m = tid >> 3, sd = tid & 7;                 // 256 = 32 rows x 8
    s_x[tid] = 0.f; s_cum[tid] = 0.f;
    out[((size_t)(brow0 + m) * T_) * SD_ + sd] = 0.f;  // cumsum[t=0] = 0
  }
  __syncthreads();

  v8f cst[2][MT_];  // c state: this wave owns hidden cols [w*32, w*32+32)
  #pragma unroll
  for (int nl = 0; nl < 2; ++nl)
    #pragma unroll
    for (int mt = 0; mt < MT_; ++mt)
      #pragma unroll
      for (int r = 0; r < 8; ++r) cst[nl][mt][r] = 0.f;

  unsigned long long fru = (unsigned long long)(uintptr_t)frags;

  for (int t = 0; t < T_ - 1; ++t) {
    const int rb = t & 1, wb = rb ^ 1;
    // Launder the base address in an SGPR: loop-variant => no hoist/spill.
    asm volatile("" : "+s"(fru));
    gcHalf* fr = reinterpret_cast<gcHalf*>(fru);   // AS(1): global_load form

    // ---- 1) xt = concat(x, noise_t), f16, K padded to 64 ----
    for (int i = tid; i < ROWS_ * 64; i += 256) {
      int m = i >> 6, c = i & 63;
      float v = 0.f;
      if (c < SD_)            v = s_x[m * SD_ + c];
      else if (c < SD_ + ND_) v = noise[((size_t)(brow0 + m) * T_ + t) * ND_ + (c - SD_)];
      s_xt[i] = (_Float16)v;
    }
    __syncthreads();

    // ---- 2) gates: z = xt@Wx + h@Wh + b ; LSTM update ----
    #pragma unroll
    for (int nl = 0; nl < 2; ++nl) {
      const int ncol = w * 32 + nl * 16;  // within-gate column base
      const int nt   = ncol >> 4;         // within-gate tile 0..15
      v8f acc[4][MT_];
      #pragma unroll
      for (int g = 0; g < 4; ++g)
        #pragma unroll
        for (int mt = 0; mt < MT_; ++mt)
          #pragma unroll
          for (int r = 0; r < 8; ++r) acc[g][mt][r] = 0.f;

      // Unified 10-chunk K loop (2 x Wx, 8 x Wh), B frags prefetched 1 ahead.
      auto fidx = [&](int q, int g) {
        return q < 2 ? OFF_WX + (g * 16 + nt) * 2 + q
                     : OFF_WH + (g * 16 + nt) * 8 + (q - 2);
      };
      v16h bf[4], bfn[4];
      #pragma unroll
      for (int g = 0; g < 4; ++g) bf[g] = load_b(fr, fidx(0, g), lane);
      #pragma unroll
      for (int q = 0; q < 10; ++q) {
        if (q < 9) {
          #pragma unroll
          for (int g = 0; g < 4; ++g) bfn[g] = load_b(fr, fidx(q + 1, g), lane);
        }
        v16h a0, a1;
        if (q < 2) {
          a0 = load_a(s_xt,           64, q, lane);
          a1 = load_a(s_xt + 16 * 64, 64, q, lane);
        } else {
          a0 = load_a(s_h[rb],            256, q - 2, lane);
          a1 = load_a(s_h[rb] + 16 * 256, 256, q - 2, lane);
        }
        #pragma unroll
        for (int g = 0; g < 4; ++g) {
          acc[g][0] = wmma16(a0, bf[g], acc[g][0]);
          acc[g][1] = wmma16(a1, bf[g], acc[g][1]);
        }
        #pragma unroll
        for (int g = 0; g < 4; ++g) bf[g] = bfn[g];
      }

      const int n = ncol + bm;  // hidden column 0..255
      float bi = b4[0 * H_ + n], bff = b4[1 * H_ + n];
      float bg = b4[2 * H_ + n], bo  = b4[3 * H_ + n];
      #pragma unroll
      for (int mt = 0; mt < MT_; ++mt) {
        #pragma unroll
        for (int r = 0; r < 8; ++r) {
          float iv = sigm_(acc[0][mt][r] + bi);
          float fv = sigm_(acc[1][mt][r] + bff);
          float gv = tanh_(acc[2][mt][r] + bg);
          float ov = sigm_(acc[3][mt][r] + bo);
          float cn = fv * cst[nl][mt][r] + iv * gv;
          cst[nl][mt][r] = cn;
          float hv = ov * tanh_(cn);
          int m = mt * 16 + r + (ch << 3);
          s_h[wb][m * 256 + n] = (_Float16)hv;
        }
      }
    }
    __syncthreads();

    // ---- 3) MLP layer 1 (mu + var nets: 32 N-tiles / 8 waves) ----
    #pragma unroll
    for (int tt = 0; tt < 4; ++tt) {
      int tgl = w * 4 + tt;
      int net = tgl >> 4;   // 0 = mu, 1 = var
      int nt  = tgl & 15;
      int base = (net ? OFF_WV1 : OFF_WM1) + nt * 8;
      v8f acc[MT_] = {{}, {}};
      v16h bf = load_b(fr, base, lane), bfn;
      #pragma unroll
      for (int kc = 0; kc < 8; ++kc) {
        if (kc < 7) bfn = load_b(fr, base + kc + 1, lane);
        v16h a0 = load_a(s_h[wb],            256, kc, lane);
        v16h a1 = load_a(s_h[wb] + 16 * 256, 256, kc, lane);
        acc[0] = wmma16(a0, bf, acc[0]);
        acc[1] = wmma16(a1, bf, acc[1]);
        bf = bfn;
      }
      int n = nt * 16 + bm;
      float bb = net ? bv1[n] : bm1[n];
      #pragma unroll
      for (int mt = 0; mt < MT_; ++mt)
        #pragma unroll
        for (int r = 0; r < 8; ++r) {
          float v = acc[mt][r] + bb; v = v > 0.f ? v : 0.f;  // relu
          s_a1[net][(mt * 16 + r + (ch << 3)) * 256 + n] = (_Float16)v;
        }
    }
    __syncthreads();

    // ---- 4) MLP layer 2 ----
    #pragma unroll
    for (int tt = 0; tt < 4; ++tt) {
      int tgl = w * 4 + tt;
      int net = tgl >> 4;
      int nt  = tgl & 15;
      int base = (net ? OFF_WV2 : OFF_WM2) + nt * 8;
      v8f acc[MT_] = {{}, {}};
      v16h bf = load_b(fr, base, lane), bfn;
      #pragma unroll
      for (int kc = 0; kc < 8; ++kc) {
        if (kc < 7) bfn = load_b(fr, base + kc + 1, lane);
        v16h a0 = load_a(s_a1[net],            256, kc, lane);
        v16h a1 = load_a(s_a1[net] + 16 * 256, 256, kc, lane);
        acc[0] = wmma16(a0, bf, acc[0]);
        acc[1] = wmma16(a1, bf, acc[1]);
        bf = bfn;
      }
      int n = nt * 16 + bm;
      float bb = net ? bv2[n] : bm2[n];
      #pragma unroll
      for (int mt = 0; mt < MT_; ++mt)
        #pragma unroll
        for (int r = 0; r < 8; ++r) {
          float v = acc[mt][r] + bb; v = v > 0.f ? v : 0.f;
          s_a2[net][(mt * 16 + r + (ch << 3)) * 256 + n] = (_Float16)v;
        }
    }
    __syncthreads();

    // ---- 5) MLP layer 3 (N=8 padded to 16; waves 0/1 = mu/var) ----
    if (w < 2) {
      int net = w;
      int base = net ? OFF_WV3 : OFF_WM3;
      v8f acc[MT_] = {{}, {}};
      v16h bf = load_b(fr, base, lane), bfn;
      #pragma unroll
      for (int kc = 0; kc < 8; ++kc) {
        if (kc < 7) bfn = load_b(fr, base + kc + 1, lane);
        v16h a0 = load_a(s_a2[net],            256, kc, lane);
        v16h a1 = load_a(s_a2[net] + 16 * 256, 256, kc, lane);
        acc[0] = wmma16(a0, bf, acc[0]);
        acc[1] = wmma16(a1, bf, acc[1]);
        bf = bfn;
      }
      if (bm < SD_) {
        float bb = net ? bv3[bm] : bm3[bm];
        #pragma unroll
        for (int mt = 0; mt < MT_; ++mt)
          #pragma unroll
          for (int r = 0; r < 8; ++r) {
            float v = acc[mt][r] + bb;
            int m = mt * 16 + r + (ch << 3);
            if (net) s_lv[m * SD_ + bm] = v; else s_mu[m * SD_ + bm] = v;
          }
      }
    }
    __syncthreads();

    // ---- 6) reparameterized sample + running cumsum + output ----
    {
      int m = tid >> 3, sd = tid & 7;                 // all 256 threads
      size_t brow = (size_t)(brow0 + m);
      float e  = eps[(brow * (T_ - 1) + t) * SD_ + sd];
      float xv = s_mu[tid] + __expf(0.5f * s_lv[tid]) * DT_ * e;
      s_x[tid] = xv;
      float cu = s_cum[tid] + xv;
      s_cum[tid] = cu;
      out[(brow * T_ + (t + 1)) * SD_ + sd] = cu;
    }
    __syncthreads();
  }
}

// ---------------------------------------------------------------------------
extern "C" void kernel_launch(void* const* d_in, const int* in_sizes, int n_in,
                              void* d_out, int out_size, void* d_ws,
                              size_t ws_size, hipStream_t stream) {
  (void)in_sizes; (void)n_in; (void)out_size; (void)ws_size;
  const float* noise = (const float*)d_in[0];
  const float* eps   = (const float*)d_in[1];
  const float* Wx    = (const float*)d_in[2];
  const float* Wh    = (const float*)d_in[3];
  const float* b4    = (const float*)d_in[4];
  const float* Wm1   = (const float*)d_in[5];
  const float* bm1   = (const float*)d_in[6];
  const float* Wm2   = (const float*)d_in[7];
  const float* bm2   = (const float*)d_in[8];
  const float* Wm3   = (const float*)d_in[9];
  const float* bm3   = (const float*)d_in[10];
  const float* Wv1   = (const float*)d_in[11];
  const float* bv1   = (const float*)d_in[12];
  const float* Wv2   = (const float*)d_in[13];
  const float* bv2   = (const float*)d_in[14];
  const float* Wv3   = (const float*)d_in[15];
  const float* bv3   = (const float*)d_in[16];

  unsigned*  ws_u = (unsigned*)d_ws;
  _Float16*  frp  = (_Float16*)d_ws;

  auto pack = [&](const float* W, int K, int N, int nKc, int nNt, int off) {
    int total = nKc * nNt * 256;
    pack_b_frags<<<(total + 255) / 256, 256, 0, stream>>>(
        W, K, N, nKc, nNt, ws_u + (size_t)off * 256);
  };
  pack(Wx,  SD_ + ND_, 4 * H_, 2, 64, OFF_WX);
  pack(Wh,  H_,        4 * H_, 8, 64, OFF_WH);
  pack(Wm1, H_, H_,  8, 16, OFF_WM1);
  pack(Wm2, H_, H_,  8, 16, OFF_WM2);
  pack(Wm3, H_, SD_, 8, 1,  OFF_WM3);
  pack(Wv1, H_, H_,  8, 16, OFF_WV1);
  pack(Wv2, H_, H_,  8, 16, OFF_WV2);
  pack(Wv3, H_, SD_, 8, 1,  OFF_WV3);

  genlstm_seq<<<B_ / ROWS_, 256, 0, stream>>>(noise, eps, b4, bm1, bm2, bm3,
                                              bv1, bv2, bv3, frp,
                                              (float*)d_out);
}
